// MultiHeadAttention_4535485464712
// MI455X (gfx1250) — matrix operand order repaired and checked by reading, not compiled, not run on hardware
//
#include <hip/hip_runtime.h>
#include <hip/hip_bf16.h>

// ---------------------------------------------------------------------------
// MHA forward for B=2, T=2048, D=1024, H=16, DK=64 on gfx1250 (wave32, WMMA).
// d_out = [B*T*D] fp32 output  ++  [B*H*T*T] fp32 attention probabilities.
// bf16 V_WMMA_F32_16X16X32_BF16, fp32 accumulation.
// Staging via GLOBAL_LOAD_ASYNC_TO_LDS_B128 (ASYNCcnt), double-buffered in
// the projection GEMM so DMA overlaps WMMA.  attn traffic is streaming
// (512MB >> 192MB L2) so all attn accesses are non-temporal.
// ---------------------------------------------------------------------------

typedef __attribute__((ext_vector_type(16))) __bf16 v16bf;
typedef __attribute__((ext_vector_type(8)))  float  v8f;
typedef __attribute__((ext_vector_type(4)))  float  f32x4;

struct alignas(16) U4 { unsigned int x, y, z, w; };

union Frag16 {
  v16bf v;
  unsigned int u[8];
  unsigned short s[16];
};

__device__ __forceinline__ void setq(Frag16& f, int half, const U4& d) {
  f.u[half * 4 + 0] = d.x;
  f.u[half * 4 + 1] = d.y;
  f.u[half * 4 + 2] = d.z;
  f.u[half * 4 + 3] = d.w;
}

__device__ __forceinline__ unsigned short f2bf(float f) {
  unsigned int u = __float_as_uint(f);
  u += 0x7FFFu + ((u >> 16) & 1u);   // round-to-nearest-even
  return (unsigned short)(u >> 16);
}

__device__ __forceinline__ unsigned int pk2(unsigned short a, unsigned short b) {
  return (unsigned int)a | ((unsigned int)b << 16);
}

// Async DMA: 16 bytes per lane, global -> LDS.  Tracked by ASYNCcnt.
__device__ __forceinline__ void async_ld_b128(U4* lds_dst, const void* gsrc) {
  unsigned loff = (unsigned)(unsigned long long)(uintptr_t)lds_dst; // LDS byte offset
  unsigned long long g = (unsigned long long)(uintptr_t)gsrc;
  asm volatile("global_load_async_to_lds_b128 %0, %1, off"
               :: "v"(loff), "v"(g) : "memory");
}
__device__ __forceinline__ void wait_async0() {
  asm volatile("s_wait_asynccnt 0" ::: "memory");
}
// Allow the most recent 6 async ops (next stage's prefetch) to stay in flight.
__device__ __forceinline__ void wait_async6() {
  asm volatile("s_wait_asynccnt 6" ::: "memory");
}

// ---------------------------------------------------------------------------
// fp32 -> bf16 elementwise conversion
// ---------------------------------------------------------------------------
__global__ __launch_bounds__(256) void mha_cvt_bf16(
    const float* __restrict__ in, unsigned short* __restrict__ out, int n) {
  int i = blockIdx.x * 1024 + threadIdx.x;
#pragma unroll
  for (int j = 0; j < 4; ++j) {
    int idx = i + j * 256;
    if (idx < n) out[idx] = f2bf(__builtin_nontemporal_load(in + idx));
  }
}

// ---------------------------------------------------------------------------
// C[M,1024] = (A[M,1024] @ W[1024,1024]^T + bias) * scale
// Block tile 128x64, K-stage 64, double-buffered async LDS staging so the
// next stage's DMA overlaps the current stage's 8 WMMAs.  8 waves, each wave
// 16 rows x 64 cols (4 accumulators).  A, W bf16.
// ---------------------------------------------------------------------------
__global__ __launch_bounds__(256) void mha_gemm_nt(
    const unsigned short* __restrict__ A,
    const unsigned short* __restrict__ W,
    const float* __restrict__ bias,
    float* __restrict__ Cf,
    unsigned short* __restrict__ Cbf,
    int M, float scale) {
  const int KD = 1024, ND = 1024;
  const int NSTAGE = 16;            // 1024 / 64
  __shared__ U4 sA[2][128 * 8];     // 2 x (128 rows x 64 k bf16) = 32KB
  __shared__ U4 sB[2][64 * 8];      // 2 x (64 rows x 64 k bf16)  = 16KB

  const int t = threadIdx.x;
  const int wave = t >> 5, lane = t & 31;
  const int l15 = lane & 15, kg = lane >> 4;
  const int rowBase = blockIdx.x * 128;
  const int colBase = blockIdx.y * 64;

  const int arow = t >> 1, aoff = (t & 1) * 32;   // 32 elems = 4 x b128
  const int brow = t >> 2, boff = (t & 3) * 16;   // 16 elems = 2 x b128

  auto issue = [&](int k0, int buf) {
    const U4* ga = reinterpret_cast<const U4*>(
        A + (size_t)(rowBase + arow) * KD + k0 + aoff);
#pragma unroll
    for (int j = 0; j < 4; ++j)
      async_ld_b128(&sA[buf][arow * 8 + (aoff >> 3) + j], ga + j);
    const U4* gb = reinterpret_cast<const U4*>(
        W + (size_t)(colBase + brow) * KD + k0 + boff);
#pragma unroll
    for (int j = 0; j < 2; ++j)
      async_ld_b128(&sB[buf][brow * 8 + (boff >> 3) + j], gb + j);
  };

  v8f acc[4] = {};

  issue(0, 0);
  for (int s = 0; s < NSTAGE; ++s) {
    const int cur = s & 1;
    __syncthreads();                 // everyone done reading buffer cur^1
    if (s + 1 < NSTAGE) {
      issue((s + 1) * 64, cur ^ 1);  // prefetch next stage under compute
      wait_async6();                 // current stage's 6 copies complete
    } else {
      wait_async0();
    }
    __syncthreads();                 // current stage visible to all waves

#pragma unroll
    for (int kk = 0; kk < 64; kk += 32) {
      Frag16 af;
      int m = wave * 16 + l15;
      setq(af, 0, sA[cur][m * 8 + (kk >> 3) + kg]);
      setq(af, 1, sA[cur][m * 8 + (kk >> 3) + 2 + kg]);
#pragma unroll
      for (int c = 0; c < 4; ++c) {
        Frag16 bf;
        int n = c * 16 + l15;
        setq(bf, 0, sB[cur][n * 8 + (kk >> 3) + kg]);
        setq(bf, 1, sB[cur][n * 8 + (kk >> 3) + 2 + kg]);
        acc[c] = __builtin_amdgcn_wmma_f32_16x16x32_bf16(
            false, af.v, false, bf.v, (short)0, acc[c], false, false);
      }
    }
  }

#pragma unroll
  for (int c = 0; c < 4; ++c) {
    int col = colBase + c * 16 + l15;
    float bv = bias[col];
#pragma unroll
    for (int i = 0; i < 8; ++i) {
      int row = rowBase + wave * 16 + i + kg * 8;
      float v = (acc[c][i] + bv) * scale;
      size_t idx = (size_t)row * ND + col;
      if (Cf)  Cf[idx] = v;
      if (Cbf) Cbf[idx] = f2bf(v);
    }
  }
}

// ---------------------------------------------------------------------------
// scores[z, q, k] = sum_d Qs[b,q,h,d] * K[b,k,h,d]  (Q pre-scaled by 1/8)
// masked to -1e9 for k > q.  Block tile: 64 q x 128 k, z = b*H + h.
// ---------------------------------------------------------------------------
__global__ __launch_bounds__(256) void mha_scores(
    const unsigned short* __restrict__ Qbf,
    const unsigned short* __restrict__ Kbf,
    float* __restrict__ attn,
    int T, int D, int H) {
  __shared__ U4 sQ[64 * 8];    // 64 q x 64 d
  __shared__ U4 sK[128 * 8];   // 128 k x 64 d

  const int t = threadIdx.x;
  const int wave = t >> 5, lane = t & 31;
  const int l15 = lane & 15, kg = lane >> 4;
  const int kBase = blockIdx.x * 128;
  const int qBase = blockIdx.y * 64;
  const int z = blockIdx.z;
  const int b = z / H, h = z % H;

  if (kBase > qBase + 63) {  // fully masked tile: stream -1e9 (NT, f32x4)
    const f32x4 neg = {-1e9f, -1e9f, -1e9f, -1e9f};
    for (int i = t; i < 64 * 32; i += 256) {       // 64 rows x 32 f32x4
      int r = qBase + (i >> 5);
      int c = kBase + (i & 31) * 4;
      __builtin_nontemporal_store(
          neg, (f32x4*)(attn + ((size_t)z * T + r) * T + c));
    }
    return;
  }

  {
    int qr = t >> 2, qo = (t & 3) * 16;
    const U4* gq = reinterpret_cast<const U4*>(
        Qbf + (size_t)(b * T + qBase + qr) * D + h * 64 + qo);
#pragma unroll
    for (int j = 0; j < 2; ++j)
      async_ld_b128(&sQ[qr * 8 + (qo >> 3) + j], gq + j);
    int kr = t >> 1, ko = (t & 1) * 32;
    const U4* gk = reinterpret_cast<const U4*>(
        Kbf + (size_t)(b * T + kBase + kr) * D + h * 64 + ko);
#pragma unroll
    for (int j = 0; j < 4; ++j)
      async_ld_b128(&sK[kr * 8 + (ko >> 3) + j], gk + j);
  }
  wait_async0();
  __syncthreads();

  const int wq = wave >> 1, wn = wave & 1;
  v8f acc[4] = {};
#pragma unroll
  for (int kk = 0; kk < 64; kk += 32) {
    Frag16 af;
    int m = wq * 16 + l15;
    setq(af, 0, sQ[m * 8 + (kk >> 3) + kg]);
    setq(af, 1, sQ[m * 8 + (kk >> 3) + 2 + kg]);
#pragma unroll
    for (int c = 0; c < 4; ++c) {
      Frag16 bf;
      int n = wn * 64 + c * 16 + l15;
      setq(bf, 0, sK[n * 8 + (kk >> 3) + kg]);
      setq(bf, 1, sK[n * 8 + (kk >> 3) + 2 + kg]);
      acc[c] = __builtin_amdgcn_wmma_f32_16x16x32_bf16(
          false, af.v, false, bf.v, (short)0, acc[c], false, false);
    }
  }

#pragma unroll
  for (int c = 0; c < 4; ++c) {
    int col = kBase + wn * 64 + c * 16 + l15;
#pragma unroll
    for (int i = 0; i < 8; ++i) {
      int row = qBase + wq * 16 + i + kg * 8;
      float v = (col <= row) ? acc[c][i] : -1e9f;
      __builtin_nontemporal_store(v, attn + ((size_t)z * T + row) * T + col);
    }
  }
}

// ---------------------------------------------------------------------------
// In-place row softmax over T=2048.  One block per row; each thread owns 8
// contiguous floats (two f32x4 NT accesses); LDS tree reductions.
// ---------------------------------------------------------------------------
__global__ __launch_bounds__(256) void mha_softmax(float* __restrict__ attn, int T) {
  __shared__ float red[256];
  const int t = threadIdx.x;
  float* row = attn + (size_t)blockIdx.x * T;

  f32x4 a = __builtin_nontemporal_load((const f32x4*)(row + t * 8));
  f32x4 c = __builtin_nontemporal_load((const f32x4*)(row + t * 8 + 4));

  float m = fmaxf(fmaxf(fmaxf(a.x, a.y), fmaxf(a.z, a.w)),
                  fmaxf(fmaxf(c.x, c.y), fmaxf(c.z, c.w)));
  red[t] = m;
  __syncthreads();
  for (int s = 128; s > 0; s >>= 1) {
    if (t < s) red[t] = fmaxf(red[t], red[t + s]);
    __syncthreads();
  }
  m = red[0];
  __syncthreads();

  a.x = __expf(a.x - m); a.y = __expf(a.y - m);
  a.z = __expf(a.z - m); a.w = __expf(a.w - m);
  c.x = __expf(c.x - m); c.y = __expf(c.y - m);
  c.z = __expf(c.z - m); c.w = __expf(c.w - m);
  float sum = (a.x + a.y) + (a.z + a.w) + (c.x + c.y) + (c.z + c.w);
  red[t] = sum;
  __syncthreads();
  for (int s = 128; s > 0; s >>= 1) {
    if (t < s) red[t] += red[t + s];
    __syncthreads();
  }
  float inv = 1.0f / red[0];

  a.x *= inv; a.y *= inv; a.z *= inv; a.w *= inv;
  c.x *= inv; c.y *= inv; c.z *= inv; c.w *= inv;
  __builtin_nontemporal_store(a, (f32x4*)(row + t * 8));
  __builtin_nontemporal_store(c, (f32x4*)(row + t * 8 + 4));
}

// ---------------------------------------------------------------------------
// ctx[b,q,h,d] = sum_k P[z,q,k] * V[b,k,h,d].  Block: 128 q rows, full DK=64.
// P converted fp32->bf16 while staging into LDS; k-loop stops at causal edge.
// ---------------------------------------------------------------------------
__global__ __launch_bounds__(256) void mha_attn_v(
    const float* __restrict__ attn,
    const unsigned short* __restrict__ Vbf,
    unsigned short* __restrict__ ctx,
    int T, int D, int H) {
  __shared__ U4 sP[128 * 4];  // 128 q x 32 k (bf16)
  __shared__ U4 sV[32 * 8];   // 32 k x 64 d (bf16), row-major [k][d]

  const int t = threadIdx.x;
  const int wave = t >> 5, lane = t & 31;
  const int l15 = lane & 15, kg = lane >> 4;
  const int qBase = blockIdx.x * 128;
  const int z = blockIdx.z;
  const int b = z / H, h = z % H;

  const int pr = t >> 1, po = (t & 1) * 16;
  const int vr = t >> 3, vo = (t & 7) * 8;
  const unsigned short* sVu = reinterpret_cast<const unsigned short*>(sV);

  v8f acc[4] = {};

  int kEnd = qBase + 128;          // probs are exactly 0 beyond the diagonal
  if (kEnd > T) kEnd = T;

  for (int k0 = 0; k0 < kEnd; k0 += 32) {
    {
      // V tile: pure bf16 copy -> async DMA
      const U4* vp = reinterpret_cast<const U4*>(
          Vbf + (size_t)(b * T + k0 + vr) * D + h * 64 + vo);
      async_ld_b128(&sV[vr * 8 + (vo >> 3)], vp);

      // P tile: fp32 -> bf16 conversion while staging (f32x4 NT loads)
      const float* pp = attn + ((size_t)z * T + qBase + pr) * T + k0 + po;
      f32x4 p0 = __builtin_nontemporal_load((const f32x4*)(pp + 0));
      f32x4 p1 = __builtin_nontemporal_load((const f32x4*)(pp + 4));
      f32x4 p2 = __builtin_nontemporal_load((const f32x4*)(pp + 8));
      f32x4 p3 = __builtin_nontemporal_load((const f32x4*)(pp + 12));
      U4 d0, d1;
      d0.x = pk2(f2bf(p0.x), f2bf(p0.y));
      d0.y = pk2(f2bf(p0.z), f2bf(p0.w));
      d0.z = pk2(f2bf(p1.x), f2bf(p1.y));
      d0.w = pk2(f2bf(p1.z), f2bf(p1.w));
      d1.x = pk2(f2bf(p2.x), f2bf(p2.y));
      d1.y = pk2(f2bf(p2.z), f2bf(p2.w));
      d1.z = pk2(f2bf(p3.x), f2bf(p3.y));
      d1.w = pk2(f2bf(p3.z), f2bf(p3.w));
      sP[pr * 4 + (po >> 3) + 0] = d0;
      sP[pr * 4 + (po >> 3) + 1] = d1;
    }
    wait_async0();
    __syncthreads();

    Frag16 af;
    {
      int m = wave * 16 + l15;
      setq(af, 0, sP[m * 4 + kg * 2]);
      setq(af, 1, sP[m * 4 + kg * 2 + 1]);
    }
#pragma unroll
    for (int c = 0; c < 4; ++c) {
      Frag16 bf;
      int n = c * 16 + l15;
#pragma unroll
      for (int e = 0; e < 16; ++e)
        bf.s[e] = sVu[(kg * 16 + e) * 64 + n];
      acc[c] = __builtin_amdgcn_wmma_f32_16x16x32_bf16(
          false, af.v, false, bf.v, (short)0, acc[c], false, false);
    }
    __syncthreads();
  }

#pragma unroll
  for (int c = 0; c < 4; ++c) {
    int d = c * 16 + l15;
#pragma unroll
    for (int i = 0; i < 8; ++i) {
      int row = qBase + wave * 16 + i + kg * 8;
      ctx[(size_t)(b * T + row) * D + h * 64 + d] = f2bf(acc[c][i]);
    }
  }
}

// ---------------------------------------------------------------------------
// Host-side launch
// ---------------------------------------------------------------------------
extern "C" void kernel_launch(void* const* d_in, const int* in_sizes, int n_in,
                              void* d_out, int out_size, void* d_ws, size_t ws_size,
                              hipStream_t stream) {
  (void)in_sizes; (void)n_in; (void)out_size; (void)ws_size;
  const int B = 2, T = 2048, D = 1024, H = 16;
  const int M = B * T;

  const float* x_q  = (const float*)d_in[0];
  const float* x_kv = (const float*)d_in[1];
  // d_in[2] = mask (causal, known statically)
  const float* Wq = (const float*)d_in[3];
  const float* bq = (const float*)d_in[4];
  const float* Wk = (const float*)d_in[5];
  const float* bk = (const float*)d_in[6];
  const float* Wv = (const float*)d_in[7];
  const float* bv = (const float*)d_in[8];
  const float* Wo = (const float*)d_in[9];
  const float* bo = (const float*)d_in[10];

  float* out  = (float*)d_out;             // [B*T, D]
  float* attn = out + (size_t)M * D;       // [B*H, T, T]

  char* ws = (char*)d_ws;
  size_t off = 0;
  auto alloc = [&](size_t bytes) {
    char* p = ws + off;
    off += (bytes + 255) & ~(size_t)255;
    return p;
  };
  unsigned short* xq_bf  = (unsigned short*)alloc((size_t)M * D * 2);
  unsigned short* xkv_bf = (unsigned short*)alloc((size_t)M * D * 2);
  unsigned short* Wq_bf  = (unsigned short*)alloc((size_t)D * D * 2);
  unsigned short* Wk_bf  = (unsigned short*)alloc((size_t)D * D * 2);
  unsigned short* Wv_bf  = (unsigned short*)alloc((size_t)D * D * 2);
  unsigned short* Wo_bf  = (unsigned short*)alloc((size_t)D * D * 2);
  unsigned short* Q_bf   = (unsigned short*)alloc((size_t)M * D * 2);
  unsigned short* K_bf   = (unsigned short*)alloc((size_t)M * D * 2);
  unsigned short* V_bf   = (unsigned short*)alloc((size_t)M * D * 2);
  unsigned short* ctx_bf = (unsigned short*)alloc((size_t)M * D * 2);

  dim3 blk(256);

  mha_cvt_bf16<<<(M * D + 1023) / 1024, blk, 0, stream>>>(x_q,  xq_bf,  M * D);
  mha_cvt_bf16<<<(M * D + 1023) / 1024, blk, 0, stream>>>(x_kv, xkv_bf, M * D);
  mha_cvt_bf16<<<(D * D + 1023) / 1024, blk, 0, stream>>>(Wq, Wq_bf, D * D);
  mha_cvt_bf16<<<(D * D + 1023) / 1024, blk, 0, stream>>>(Wk, Wk_bf, D * D);
  mha_cvt_bf16<<<(D * D + 1023) / 1024, blk, 0, stream>>>(Wv, Wv_bf, D * D);
  mha_cvt_bf16<<<(D * D + 1023) / 1024, blk, 0, stream>>>(Wo, Wo_bf, D * D);

  dim3 gproj(M / 128, D / 64, 1);
  // Q pre-scaled by 1/sqrt(DK) = 0.125
  mha_gemm_nt<<<gproj, blk, 0, stream>>>(xq_bf,  Wq_bf, bq, nullptr, Q_bf, M, 0.125f);
  mha_gemm_nt<<<gproj, blk, 0, stream>>>(xkv_bf, Wk_bf, bk, nullptr, K_bf, M, 1.0f);
  mha_gemm_nt<<<gproj, blk, 0, stream>>>(xkv_bf, Wv_bf, bv, nullptr, V_bf, M, 1.0f);

  dim3 gsc(T / 128, T / 64, B * H);
  mha_scores<<<gsc, blk, 0, stream>>>(Q_bf, K_bf, attn, T, D, H);

  mha_softmax<<<B * H * T, blk, 0, stream>>>(attn, T);

  dim3 gav(T / 128, 1, B * H);
  mha_attn_v<<<gav, blk, 0, stream>>>(attn, V_bf, ctx_bf, T, D, H);

  mha_gemm_nt<<<gproj, blk, 0, stream>>>(ctx_bf, Wo_bf, bo, out, nullptr, M, 1.0f);
}
